// RelativeSemanticLoss_1881195675678
// MI455X (gfx1250) — matrix operand order repaired and checked by reading, not compiled, not run on hardware
//
#include <hip/hip_runtime.h>

// ---------------------------------------------------------------------------
// RelativeSemanticLoss fused kernel for MI455X (gfx1250, wave32, WMMA + TDM)
//   sim = (pred . cls^T)/TEMP ; loss = mean over valid px of NLL(log_softmax)
// GEMM M=262144 px, N=150 cls (pad 160), K=512, bf16 WMMA, f32 accum.
// B tile staged by the Tensor Data Mover (HW row-padding -> pitch-72 LDS),
// issued early so the DMA overlaps the A-staging transpose; A staged with
// scalar-base + 32-bit-offset loads and packed bf16 LDS stores.
// ---------------------------------------------------------------------------

typedef __bf16        v16bf __attribute__((ext_vector_type(16)));
typedef float         v8f   __attribute__((ext_vector_type(8)));
typedef unsigned int  u32x4 __attribute__((ext_vector_type(4)));
typedef unsigned int  u32x8 __attribute__((ext_vector_type(8)));
typedef int           i32x4 __attribute__((ext_vector_type(4)));
typedef int           i32x8 __attribute__((ext_vector_type(8)));

#define HW        65536      // 256*256
#define DDIM      512
#define NUMCLS    150
#define NPAD      160
#define NT        10         // N tiles of 16
#define M_TILE    128        // pixels per workgroup
#define KC        64         // K chunk staged in LDS
#define PITCH     72         // LDS row pitch in bf16 elements
#define INV_TEMP  (1.0f / 0.07f)
#define IGNORE_L  255

__device__ __forceinline__ unsigned short f2bf(float f) {
  unsigned int x = __builtin_bit_cast(unsigned int, f);
  x += 0x7FFFu + ((x >> 16) & 1u);               // round-to-nearest-even
  return (unsigned short)(x >> 16);
}

// Build a 16-element bf16 WMMA fragment from two 16-byte LDS reads.
__device__ __forceinline__ v16bf ldfrag(const void* p0, const void* p1) {
  u32x4 lo = *(const u32x4*)p0;
  u32x4 hi = *(const u32x4*)p1;
  u32x8 w  = __builtin_shufflevector(lo, hi, 0, 1, 2, 3, 4, 5, 6, 7);
  return __builtin_bit_cast(v16bf, w);
}

__global__ void __launch_bounds__(256)
rsl_main(const float* __restrict__ pred, const float* __restrict__ cls,
         const unsigned short* __restrict__ clsbf,
         const int* __restrict__ idxmap, const int* __restrict__ lut,
         float* __restrict__ accum, int useTdm) {
  __shared__ __align__(16) unsigned int   AbufW[M_TILE * PITCH / 2]; // 18 KB
  __shared__ __align__(16) unsigned short Bbuf[NPAD * PITCH];        // 22.5 KB
  const unsigned short* Abuf = (const unsigned short*)AbufW;

  const int  t    = threadIdx.x;
  const int  w    = t >> 5;        // wave id 0..7
  const int  lane = t & 31;
  const int  nc   = lane & 15;
  const bool hi   = lane >= 16;

  const int    g0p      = blockIdx.x * M_TILE;          // base global pixel
  const int    b        = g0p >> 16;                    // / HW
  const int    pimg     = g0p & (HW - 1);
  const size_t predBase = (size_t)b * DDIM * HW + pimg;

  v8f acc[NT];
#pragma unroll
  for (int n = 0; n < NT; ++n)
    acc[n] = v8f{0.f, 0.f, 0.f, 0.f, 0.f, 0.f, 0.f, 0.f};

  for (int k0 = 0; k0 < DDIM; k0 += KC) {
    // ---- issue B-tile DMA first so it overlaps the A-staging below --------
    if (useTdm && w == 0) {
      unsigned long long ga =
          (unsigned long long)(size_t)clsbf + (unsigned long long)k0 * 2ull;
      unsigned lds_off = (unsigned)(size_t)&Bbuf[0];   // low 32 = LDS offset
      u32x4 g0;
      g0[0] = 1u;                                   // count=1, user mode
      g0[1] = lds_off;                              // lds_addr
      g0[2] = (unsigned)(ga & 0xFFFFFFFFull);       // global_addr[31:0]
      g0[3] = (unsigned)((ga >> 32) & 0x1FFFFFFull) // global_addr[56:32]
              | (2u << 30);                         // type=2 (image)
      i32x8 g1;
      g1[0] = (1 << 16)      // data_size = 2 B
            | (1 << 20)      // pad_enable
            | (4 << 22)      // pad_interval: 32 DWORDs (=128 B row)
            | (3 << 25);     // pad_amount: 4 DWORDs (=16 B pad)
      g1[1] = (int)(512u << 16);   // tensor_dim0 = 512  [63:48]
      g1[2] = (int)(160u << 16);   // tensor_dim1 = 160  [95:80]
      g1[3] = (int)(64u << 16);    // tile_dim0   = 64   [127:112]
      g1[4] = 160;                 // tile_dim1   = 160  [143:128]
      g1[5] = 512;                 // tensor_dim0_stride [191:160]
      g1[6] = 0;
      g1[7] = 0;
      i32x4 z4 = {0, 0, 0, 0};
#if defined(__clang_major__) && __clang_major__ >= 23
      i32x8 z8 = {0, 0, 0, 0, 0, 0, 0, 0};
      __builtin_amdgcn_tensor_load_to_lds(g0, g1, z4, z4, z8, 0);
#else
      __builtin_amdgcn_tensor_load_to_lds(g0, g1, z4, z4, 0);
#endif
    }

    // ---- stage A: 64 channels x 128 pixels ---------------------------------
    // Uniform 64-bit base in SGPRs + 32-bit per-lane offsets (saddr+voffset
    // form) so no 64-bit address vectors are kept live across the loop.
    const float* baseK = pred + predBase + (size_t)k0 * HW;
#pragma unroll
    for (int i = 0; i < (M_TILE * KC) / (256 * 2); ++i) {   // 16 iters
      int      j   = i * 256 + t;
      int      c2  = j >> 7;                                // channel pair 0..31
      int      p   = j & 127;
      unsigned off = (unsigned)(c2 * 2 * HW + p);
      float    v0  = baseK[off];
      float    v1  = baseK[off + HW];
      if (k0 + KC < DDIM)
        __builtin_prefetch(baseK + off + (unsigned)(KC * HW), 0, 0);
      AbufW[p * (PITCH / 2) + c2] =
          (unsigned)f2bf(v0) | ((unsigned)f2bf(v1) << 16);
    }

    if (useTdm) {
      if (w == 0) __builtin_amdgcn_s_wait_tensorcnt(0);
    } else {
      // fallback: convert f32 cls in-kernel
#pragma unroll
      for (int i = 0; i < (NPAD * KC) / 256; ++i) {         // 40 iters
        int   j = i * 256 + t;
        int   r = j >> 6;
        int   c = j & 63;
        float v = (r < NUMCLS) ? cls[r * DDIM + k0 + c] : 0.0f;
        Bbuf[r * PITCH + c] = f2bf(v);
      }
    }
    __syncthreads();

    // ---- WMMA: batch B-fragment loads so one dscnt wait covers a group ----
    const int arow = (w * 16 + nc) * PITCH;
#pragma unroll
    for (int kk = 0; kk < KC; kk += 32) {
      int   aoff = arow + kk + (hi ? 8 : 0);
      v16bf a    = ldfrag(&Abuf[aoff], &Abuf[aoff + 16]);
#pragma unroll
      for (int g = 0; g < NT; g += 5) {
        v16bf bfr[5];
#pragma unroll
        for (int q = 0; q < 5; ++q) {
          int n    = g + q;
          int boff = (n * 16 + nc) * PITCH + kk + (hi ? 8 : 0);
          bfr[q]   = ldfrag(&Bbuf[boff], &Bbuf[boff + 16]);
        }
#pragma unroll
        for (int q = 0; q < 5; ++q)
          acc[g + q] = __builtin_amdgcn_wmma_f32_16x16x32_bf16(
              false, a, false, bfr[q], (short)0, acc[g + q], false, false);
      }
    }
    __syncthreads();
  }

  // ---- fused log-softmax + NLL epilogue -----------------------------------
  float s_nll = 0.0f, s_cnt = 0.0f;
#pragma unroll
  for (int j = 0; j < 8; ++j) {
    int m     = j + (hi ? 8 : 0);
    int pix   = g0p + w * 16 + m;
    int label = lut[idxmap[pix]];

    float lmax = -INFINITY;
#pragma unroll
    for (int n = 0; n < NT; ++n) {
      float l     = acc[n][j] * INV_TEMP;
      bool  valid = (n < 9) | (nc < 6);                 // class < 150
      lmax = valid ? fmaxf(lmax, l) : lmax;
    }
    lmax = fmaxf(lmax, __shfl_xor(lmax, 1, 32));
    lmax = fmaxf(lmax, __shfl_xor(lmax, 2, 32));
    lmax = fmaxf(lmax, __shfl_xor(lmax, 4, 32));
    lmax = fmaxf(lmax, __shfl_xor(lmax, 8, 32));

    float lsum = 0.0f, lsel = 0.0f;
#pragma unroll
    for (int n = 0; n < NT; ++n) {
      float l     = acc[n][j] * INV_TEMP;
      bool  valid = (n < 9) | (nc < 6);
      lsum += valid ? __expf(l - lmax) : 0.0f;
      lsel  = (n * 16 + nc == label) ? l : lsel;
    }
    lsum += __shfl_xor(lsum, 1, 32);
    lsum += __shfl_xor(lsum, 2, 32);
    lsum += __shfl_xor(lsum, 4, 32);
    lsum += __shfl_xor(lsum, 8, 32);
    lsel += __shfl_xor(lsel, 1, 32);
    lsel += __shfl_xor(lsel, 2, 32);
    lsel += __shfl_xor(lsel, 4, 32);
    lsel += __shfl_xor(lsel, 8, 32);

    float logZ = __logf(lsum) + lmax;
    if (nc == 0 && label != IGNORE_L) {
      s_nll += logZ - lsel;
      s_cnt += 1.0f;
    }
  }
  s_nll += __shfl_xor(s_nll, 16, 32);
  s_cnt += __shfl_xor(s_cnt, 16, 32);
  if (lane == 0) {
    atomicAdd(&accum[0], s_nll);
    atomicAdd(&accum[1], s_cnt);
  }
}

__global__ void rsl_prep(const float* __restrict__ cls,
                         unsigned short* __restrict__ clsbf) {
  int   idx = blockIdx.x * 256 + threadIdx.x;   // < 160*512
  int   r   = idx >> 9;
  int   c   = idx & 511;
  float v   = (r < NUMCLS) ? cls[r * DDIM + c] : 0.0f;
  clsbf[idx] = f2bf(v);
}

__global__ void rsl_init(float* accum) {
  accum[0] = 0.0f;
  accum[1] = 0.0f;
}

__global__ void rsl_fin(const float* __restrict__ accum,
                        float* __restrict__ out) {
  float c = accum[1];
  out[0]  = accum[0] / (c > 0.0f ? c : 1.0f);
}

extern "C" void kernel_launch(void* const* d_in, const int* in_sizes, int n_in,
                              void* d_out, int out_size, void* d_ws,
                              size_t ws_size, hipStream_t stream) {
  const float* pred   = (const float*)d_in[0];   // (4,512,256,256) f32
  const float* cls    = (const float*)d_in[1];   // (150,512) f32
  const int*   idxmap = (const int*)d_in[2];     // (4,1,256,256) i32
  const int*   lut    = (const int*)d_in[3];     // (512,) i32
  float*       out    = (float*)d_out;

  float*          accum = (float*)d_ws;                       // 8 B
  unsigned short* clsbf = (unsigned short*)((char*)d_ws + 64);

  const int total_pixels = in_sizes[2];          // B*H*W = 262144
  const int grid         = total_pixels / M_TILE;
  const int use_tdm =
      (ws_size >= (size_t)(64 + NPAD * DDIM * 2)) ? 1 : 0;

  rsl_init<<<1, 1, 0, stream>>>(accum);
  if (use_tdm)
    rsl_prep<<<(NPAD * DDIM) / 256, 256, 0, stream>>>(cls, clsbf);
  rsl_main<<<grid, 256, 0, stream>>>(pred, cls, idxmap ? clsbf : clsbf, idxmap,
                                     lut, accum, use_tdm);
  rsl_fin<<<1, 1, 0, stream>>>(accum, out);
}
